// SetAbstraction_36352603193822
// MI455X (gfx1250) — matrix operand order repaired and checked by reading, not compiled, not run on hardware
//
#include <hip/hip_runtime.h>
#include <hip/hip_bf16.h>
#include <cstdint>
#include <cstddef>

// ---------------------------------------------------------------------------
// PointNet++ SetAbstraction for MI455X (gfx1250, wave32, WMMA)
// Bandwidth-bound workload: BN+ReLU fused into WMMA B-fragment loads so each
// f16 activation tensor is written once and read once.
// ---------------------------------------------------------------------------
#define BATCH      16
#define NPTS       4096
#define KREG       1024
#define NSAMP      64
#define RADIUS2    0.16f
#define BN_EPS     1e-5f
#define S_TOTAL    (BATCH * KREG * NSAMP)   // 1,048,576 samples

typedef __attribute__((ext_vector_type(16))) _Float16 v16h;
typedef __attribute__((ext_vector_type(8)))  _Float16 v8h;
typedef __attribute__((ext_vector_type(8)))  float    v8f;

union HFrag { v16h v; v8h h[2]; };

// ReLU on a packed-f16 vector -> v_pk_max_num_f16
static __device__ inline v8h relu8(v8h x) {
#if __has_builtin(__builtin_elementwise_max)
    return __builtin_elementwise_max(x, (v8h)(_Float16)0);
#else
    #pragma unroll
    for (int e = 0; e < 8; ++e) x[e] = x[e] > (_Float16)0 ? x[e] : (_Float16)0;
    return x;
#endif
}

// ---------------------------------------------------------------------------
// 0) zero a float region (stats accumulators) — deterministic per launch
// ---------------------------------------------------------------------------
__global__ void zero_kernel(float* p, int n) {
    int i = blockIdx.x * blockDim.x + threadIdx.x;
    if (i < n) p[i] = 0.0f;
}

// ---------------------------------------------------------------------------
// 1) convert w2 (64x64) and w3 (128x64) to f16, row-major [O][C]
// ---------------------------------------------------------------------------
__global__ void cvt_w_kernel(const float* __restrict__ w2, const float* __restrict__ w3,
                             _Float16* __restrict__ wf2, _Float16* __restrict__ wf3) {
    int i = blockIdx.x * blockDim.x + threadIdx.x;
    if (i < 64 * 64)  wf2[i] = (_Float16)w2[i];
    if (i < 128 * 64) wf3[i] = (_Float16)w3[i];
}

// ---------------------------------------------------------------------------
// 2) Farthest point sampling: one 1024-thread workgroup per batch.
//    min-distance array in LDS; argmax via shared tree reduction with
//    lowest-index tie-break (matches jnp.argmax).  Writes centroids [b][k][3].
// ---------------------------------------------------------------------------
__global__ __launch_bounds__(1024) void fps_kernel(const float* __restrict__ pc,
                                                   float* __restrict__ centroids) {
    __shared__ float md[NPTS];
    __shared__ float redD[1024];
    __shared__ int   redI[1024];
    __shared__ float cx, cy, cz;
    __shared__ int   curFar;

    const int b = blockIdx.x;
    const int t = threadIdx.x;
    const float* px = pc + (size_t)b * 3 * NPTS;
    const float* py = px + NPTS;
    const float* pz = px + 2 * NPTS;

    for (int n = t; n < NPTS; n += 1024) md[n] = 1e10f;
    if (t == 0) curFar = 0;
    __syncthreads();

    for (int it = 0; it < KREG; ++it) {
        if (t == 0) {
            int f = curFar;
            cx = px[f]; cy = py[f]; cz = pz[f];
            float* c = centroids + ((size_t)b * KREG + it) * 3;
            c[0] = cx; c[1] = cy; c[2] = cz;
        }
        __syncthreads();

        float bestD = -1.0f; int bestI = NPTS;
        for (int n = t; n < NPTS; n += 1024) {
            float dx = px[n] - cx, dy = py[n] - cy, dz = pz[n] - cz;
            float d  = dx * dx + dy * dy + dz * dz;
            float m  = fminf(md[n], d);
            md[n] = m;
            if (m > bestD || (m == bestD && n < bestI)) { bestD = m; bestI = n; }
        }
        redD[t] = bestD; redI[t] = bestI;
        __syncthreads();
        for (int off = 512; off > 0; off >>= 1) {
            if (t < off) {
                float dO = redD[t + off]; int iO = redI[t + off];
                if (dO > redD[t] || (dO == redD[t] && iO < redI[t])) {
                    redD[t] = dO; redI[t] = iO;
                }
            }
            __syncthreads();
        }
        if (t == 0) curFar = redI[0];
        __syncthreads();
    }
}

// ---------------------------------------------------------------------------
// 3) Ball query: one wave32 per (b,k) row.  Ballot-scan the 4096 points in
//    ascending order, pack the first 64 in-ball indices, pad with the first.
// ---------------------------------------------------------------------------
__global__ void ball_query_kernel(const float* __restrict__ pc,
                                  const float* __restrict__ centroids,
                                  int* __restrict__ gidx) {
    const int row  = (blockIdx.x * blockDim.x + threadIdx.x) >> 5;  // b*KREG + k
    const int lane = threadIdx.x & 31;
    if (row >= BATCH * KREG) return;

    const int b = row >> 10;
    const float* px = pc + (size_t)b * 3 * NPTS;
    const float* py = px + NPTS;
    const float* pz = px + 2 * NPTS;
    const float cx = centroids[row * 3 + 0];
    const float cy = centroids[row * 3 + 1];
    const float cz = centroids[row * 3 + 2];
    int* out = gidx + (size_t)row * NSAMP;

    int cnt = 0, first = 0;
    bool haveFirst = false;
    for (int base = 0; base < NPTS && cnt < NSAMP; base += 32) {
        const int n = base + lane;
        float dx = px[n] - cx, dy = py[n] - cy, dz = pz[n] - cz;
        bool in = (dx * dx + dy * dy + dz * dz) <= RADIUS2;
        unsigned mask = (unsigned)__ballot(in);
        if (!haveFirst && mask) { first = base + __builtin_ctz(mask); haveFirst = true; }
        int prefix = __builtin_popcount(mask & ((1u << lane) - 1u));
        int pos = cnt + prefix;
        if (in && pos < NSAMP) out[pos] = n;
        cnt += __builtin_popcount(mask);
    }
    if (cnt < NSAMP) {
        for (int i = cnt + lane; i < NSAMP; i += 32) out[i] = first;
    }
}

// ---------------------------------------------------------------------------
// 4) Layer-1 (3 -> 64) + bias, gather & center on the fly.  Writes y1 as f16
//    [s][64] (pre-norm); accumulates per-channel sum / sumsq via LDS atomics.
// ---------------------------------------------------------------------------
__global__ __launch_bounds__(256) void layer1_kernel(
    const float* __restrict__ pc, const float* __restrict__ centroids,
    const int* __restrict__ gidx,
    const float* __restrict__ w1, const float* __restrict__ b1,
    _Float16* __restrict__ y1, float* __restrict__ gSum, float* __restrict__ gSq) {
    __shared__ float sSum[64], sSq[64];
    const int t = threadIdx.x;
    if (t < 64) { sSum[t] = 0.0f; sSq[t] = 0.0f; }
    __syncthreads();

    const int s   = blockIdx.x * blockDim.x + t;   // (b*K + k)*64 + j
    const int b   = s >> 16;
    const int row = s >> 6;
    const int idx = gidx[s];
    const float* px = pc + (size_t)b * 3 * NPTS;
    const float x = px[idx]            - centroids[row * 3 + 0];
    const float y = px[NPTS + idx]     - centroids[row * 3 + 1];
    const float z = px[2 * NPTS + idx] - centroids[row * 3 + 2];

    _Float16* yo = y1 + (size_t)s * 64;
    #pragma unroll 8
    for (int o = 0; o < 64; ++o) {
        float v = b1[o] + w1[o * 3 + 0] * x + w1[o * 3 + 1] * y + w1[o * 3 + 2] * z;
        yo[o] = (_Float16)v;
        atomicAdd(&sSum[o], v);
        atomicAdd(&sSq[o], v * v);
    }
    __syncthreads();
    if (t < 64) { atomicAdd(&gSum[t], sSum[t]); atomicAdd(&gSq[t], sSq[t]); }
}

// ---------------------------------------------------------------------------
// 5) BN finalize: scale = gamma*rsqrt(var+eps), shift = beta - mean*scale
//    Emits both f32 (for maxpool) and f16 (for fused GEMM consumption).
// ---------------------------------------------------------------------------
__global__ void finalize_stats_kernel(const float* __restrict__ sum, const float* __restrict__ sq,
                                      const float* __restrict__ gamma, const float* __restrict__ beta,
                                      float* __restrict__ scale, float* __restrict__ shift,
                                      _Float16* __restrict__ scaleh, _Float16* __restrict__ shifth,
                                      int C) {
    int t = threadIdx.x;
    if (t < C) {
        const float invN = 1.0f / (float)S_TOTAL;
        float m  = sum[t] * invN;
        float v  = sq[t] * invN - m * m;
        float rs = rsqrtf(v + BN_EPS);
        float sc = rs * gamma[t];
        float sf = beta[t] - m * sc;
        scale[t]  = sc;
        shift[t]  = sf;
        scaleh[t] = (_Float16)sc;
        shifth[t] = (_Float16)sf;
    }
}

// ---------------------------------------------------------------------------
// 6) WMMA GEMM with fused input BN+ReLU:
//       Y[s][o] = bias[o] + sum_c W[o][c] * relu(X[s][c]*scale[c]+shift[c])
//    Each wave: one 16-out-ch tile x 64 samples (4 N-tiles), A fragments
//    loaded once and reused across N-tiles; v_wmma_f32_16x16x32_f16 K-loop.
//    Stats: register accumulate -> shfl_xor butterfly over the 16-lane N
//    groups -> 16 ds_add_f32 per wave -> 1 global atomic per channel/block.
//    Fragment mapping used here (wave32, same k order for A and B):
//      A lane: M=lane&15; K chunk at 16*half (16 halfs contiguous)
//      B lane: N=lane&15; K chunk at 16*half (16 halfs contiguous)
//      C/D lane: N=lane&15; VGPR v -> M = v + 8*half
// ---------------------------------------------------------------------------
template <int CIN, int COUT>
__global__ __launch_bounds__(256) void gemm_wmma_kernel(
    const _Float16* __restrict__ X, const _Float16* __restrict__ W,
    const float* __restrict__ bias,
    const _Float16* __restrict__ bnScale, const _Float16* __restrict__ bnShift,
    _Float16* __restrict__ Y,
    float* __restrict__ gSum, float* __restrict__ gSq) {
    constexpr int MT  = COUT / 16;   // M tiles per sample group
    constexpr int NG  = 8 / MT;      // sample groups per 8-wave block
    constexpr int NTW = 4;           // 16-sample tiles per wave
    __shared__ float sSum[COUT], sSq[COUT];
    const int t = threadIdx.x;
    if (t < COUT) { sSum[t] = 0.0f; sSq[t] = 0.0f; }
    __syncthreads();

    const int wave = t >> 5, lane = t & 31;
    const int mt = wave % MT, ng = wave / MT;
    const int o0 = mt * 16;
    const int half = lane >> 4;
    const int nl   = lane & 15;
    const int sBase = (blockIdx.x * NG + ng) * (16 * NTW);

    // A (weight) fragments: load once, reuse across all 4 N-tiles
    HFrag a[CIN / 32];
    const _Float16* wrow = W + (size_t)(o0 + nl) * CIN;
    #pragma unroll
    for (int kb = 0; kb < CIN / 32; ++kb) {
        a[kb].h[0] = *(const v8h*)(wrow + kb * 32 + 16 * half);
        a[kb].h[1] = *(const v8h*)(wrow + kb * 32 + 16 * half + 8);
    }
    // BN scale/shift chunks matching the B-fragment channel span (uniform/N)
    v8h sc[CIN / 32][2], sh[CIN / 32][2];
    #pragma unroll
    for (int kb = 0; kb < CIN / 32; ++kb) {
        sc[kb][0] = *(const v8h*)(bnScale + kb * 32 + 16 * half);
        sc[kb][1] = *(const v8h*)(bnScale + kb * 32 + 16 * half + 8);
        sh[kb][0] = *(const v8h*)(bnShift + kb * 32 + 16 * half);
        sh[kb][1] = *(const v8h*)(bnShift + kb * 32 + 16 * half + 8);
    }

    float accS[8], accQ[8];
    #pragma unroll
    for (int v = 0; v < 8; ++v) { accS[v] = 0.0f; accQ[v] = 0.0f; }

    #pragma unroll
    for (int nt = 0; nt < NTW; ++nt) {
        const int s0 = sBase + nt * 16;
        const _Float16* xrow = X + (size_t)(s0 + nl) * CIN;
        if (nt + 1 < NTW)   // prefetch next streaming tile -> global_prefetch_b8
            __builtin_prefetch(X + (size_t)(s0 + 16 + nl) * CIN, 0, 1);

        v8f c;
        #pragma unroll
        for (int v = 0; v < 8; ++v) c[v] = bias[o0 + v + 8 * half];

        #pragma unroll
        for (int kb = 0; kb < CIN / 32; ++kb) {
            HFrag bf;
            bf.h[0] = *(const v8h*)(xrow + kb * 32 + 16 * half);
            bf.h[1] = *(const v8h*)(xrow + kb * 32 + 16 * half + 8);
            // fused BN + ReLU: v_pk_fma_f16 + v_pk_max_num_f16
            bf.h[0] = relu8(bf.h[0] * sc[kb][0] + sh[kb][0]);
            bf.h[1] = relu8(bf.h[1] * sc[kb][1] + sh[kb][1]);
            c = __builtin_amdgcn_wmma_f32_16x16x32_f16(
                    /*neg_a=*/false, a[kb].v, /*neg_b=*/false, bf.v,
                    /*c_mod=*/(short)0, c, /*reuse_a=*/false, /*reuse_b=*/false);
        }

        // store 8 contiguous channels (16B, aligned) + register stat accum
        _Float16* yout = Y + (size_t)(s0 + nl) * COUT + o0 + 8 * half;
        v8h outh;
        #pragma unroll
        for (int v = 0; v < 8; ++v) {
            float val = c[v];
            outh[v] = (_Float16)val;
            accS[v] += val;
            accQ[v] += val * val;
        }
        *(v8h*)yout = outh;
    }

    // butterfly-reduce over the 16 lanes of each half (N dimension)
    #pragma unroll
    for (int v = 0; v < 8; ++v) {
        float s = accS[v], q = accQ[v];
        for (int m = 1; m <= 8; m <<= 1) {
            s += __shfl_xor(s, m, 32);
            q += __shfl_xor(q, m, 32);
        }
        if (nl == 0) {
            atomicAdd(&sSum[o0 + v + 8 * half], s);
            atomicAdd(&sSq[o0 + v + 8 * half], q);
        }
    }
    __syncthreads();
    if (t < COUT) { atomicAdd(&gSum[t], sSum[t]); atomicAdd(&gSq[t], sSq[t]); }
}

// ---------------------------------------------------------------------------
// 7) Max-pool over the 64-sample axis with layer-3 BN+ReLU applied on the fly
//    feats layout (B, 128, K)
// ---------------------------------------------------------------------------
__global__ void maxpool_kernel(const _Float16* __restrict__ y3,
                               const float* __restrict__ scale,
                               const float* __restrict__ shift,
                               float* __restrict__ feats) {
    const int i   = blockIdx.x * blockDim.x + threadIdx.x;  // (b*K + k)*128 + o
    const int o   = i & 127;
    const int row = i >> 7;            // b*K + k
    const int b   = row >> 10;
    const int k   = row & 1023;
    const float sc = scale[o], sh = shift[o];
    const _Float16* yp = y3 + (size_t)row * NSAMP * 128 + o;
    float m = -1e30f;
    #pragma unroll 8
    for (int j = 0; j < NSAMP; ++j)
        m = fmaxf(m, (float)yp[(size_t)j * 128] * sc + sh);
    feats[((size_t)b * 128 + o) * KREG + k] = fmaxf(m, 0.0f);
}

// ---------------------------------------------------------------------------
// 8) centroids (B,K,3) -> output (B,3,K)
// ---------------------------------------------------------------------------
__global__ void centroids_out_kernel(const float* __restrict__ centroids,
                                     float* __restrict__ out) {
    int i = blockIdx.x * blockDim.x + threadIdx.x;
    if (i >= BATCH * 3 * KREG) return;
    int k = i & 1023;
    int c = (i >> 10) % 3;
    int b = i / (3 * 1024);
    out[i] = centroids[((size_t)b * KREG + k) * 3 + c];
}

// ---------------------------------------------------------------------------
// Launch
// ---------------------------------------------------------------------------
extern "C" void kernel_launch(void* const* d_in, const int* in_sizes, int n_in,
                              void* d_out, int out_size, void* d_ws, size_t ws_size,
                              hipStream_t stream) {
    (void)in_sizes; (void)n_in; (void)out_size; (void)ws_size;
    const float* pc     = (const float*)d_in[0];
    const float* w1     = (const float*)d_in[1];
    const float* b1     = (const float*)d_in[2];
    const float* gamma1 = (const float*)d_in[3];
    const float* beta1  = (const float*)d_in[4];
    const float* w2     = (const float*)d_in[5];
    const float* b2     = (const float*)d_in[6];
    const float* gamma2 = (const float*)d_in[7];
    const float* beta2  = (const float*)d_in[8];
    const float* w3     = (const float*)d_in[9];
    const float* b3     = (const float*)d_in[10];
    const float* gamma3 = (const float*)d_in[11];
    const float* beta3  = (const float*)d_in[12];

    char* ws = (char*)d_ws;
    size_t off = 0;
    float* centroids = (float*)(ws + off); off += (size_t)BATCH * KREG * 3 * sizeof(float);
    int*   gidx      = (int*)(ws + off);   off += (size_t)S_TOTAL * sizeof(int);
    float* stats     = (float*)(ws + off); off += 1024 * sizeof(float);
    // stats float layout:
    float* sum1 = stats +   0, *sq1 = stats +  64, *scale1 = stats + 128, *shift1 = stats + 192;
    float* sum2 = stats + 256, *sq2 = stats + 320, *scale2 = stats + 384, *shift2 = stats + 448;
    float* sum3 = stats + 512, *sq3 = stats + 640, *scale3 = stats + 768, *shift3 = stats + 896;
    _Float16* statsH  = (_Float16*)(ws + off); off += 512 * sizeof(_Float16);
    _Float16* scale1h = statsH +   0, *shift1h = statsH +  64;
    _Float16* scale2h = statsH + 128, *shift2h = statsH + 192;
    _Float16* scale3h = statsH + 256, *shift3h = statsH + 384;   // unused by GEMM
    _Float16* wf2 = (_Float16*)(ws + off); off += 64 * 64 * sizeof(_Float16);
    _Float16* wf3 = (_Float16*)(ws + off); off += 128 * 64 * sizeof(_Float16);
    off = (off + 255) & ~(size_t)255;
    _Float16* y1 = (_Float16*)(ws + off);  off += (size_t)S_TOTAL * 64 * sizeof(_Float16);
    _Float16* y2 = (_Float16*)(ws + off);  off += (size_t)S_TOTAL * 64 * sizeof(_Float16);
    _Float16* y3 = (_Float16*)(ws + off);  off += (size_t)S_TOTAL * 128 * sizeof(_Float16);

    float* outCentroids = (float*)d_out;                       // B*3*K
    float* outFeats     = (float*)d_out + BATCH * 3 * KREG;    // B*128*K

    // 0) zero stats accumulators (deterministic per call)
    zero_kernel<<<4, 256, 0, stream>>>(stats, 1024);
    // 1) f16 weights
    cvt_w_kernel<<<32, 256, 0, stream>>>(w2, w3, wf2, wf3);
    // 2) FPS
    fps_kernel<<<BATCH, 1024, 0, stream>>>(pc, centroids);
    // 3) ball query: 16384 rows, 8 waves/block
    ball_query_kernel<<<(BATCH * KREG) / 8, 256, 0, stream>>>(pc, centroids, gidx);
    // 4) layer 1 (+stats1)
    layer1_kernel<<<S_TOTAL / 256, 256, 0, stream>>>(pc, centroids, gidx, w1, b1, y1, sum1, sq1);
    // 5) BN1 finalize (f32 + f16)
    finalize_stats_kernel<<<1, 64, 0, stream>>>(sum1, sq1, gamma1, beta1,
                                                scale1, shift1, scale1h, shift1h, 64);
    // 6) layer 2 GEMM via WMMA, BN1+ReLU fused on input (+stats2)
    gemm_wmma_kernel<64, 64><<<S_TOTAL / 128, 256, 0, stream>>>(
        y1, wf2, b2, scale1h, shift1h, y2, sum2, sq2);
    // 7) BN2 finalize
    finalize_stats_kernel<<<1, 64, 0, stream>>>(sum2, sq2, gamma2, beta2,
                                                scale2, shift2, scale2h, shift2h, 64);
    // 8) layer 3 GEMM via WMMA, BN2+ReLU fused on input (+stats3)
    gemm_wmma_kernel<64, 128><<<S_TOTAL / 64, 256, 0, stream>>>(
        y2, wf3, b3, scale2h, shift2h, y3, sum3, sq3);
    // 9) BN3 finalize; max-pool fuses BN3+ReLU
    finalize_stats_kernel<<<1, 128, 0, stream>>>(sum3, sq3, gamma3, beta3,
                                                 scale3, shift3, scale3h, shift3h, 128);
    maxpool_kernel<<<(BATCH * KREG * 128) / 256, 256, 0, stream>>>(y3, scale3, shift3, outFeats);
    // 10) centroids transpose to (B,3,K)
    centroids_out_kernel<<<(BATCH * 3 * KREG + 255) / 256, 256, 0, stream>>>(centroids, outCentroids);
}
